// SBFTransformerV2_16183436772066
// MI455X (gfx1250) — matrix operand order repaired and checked by reading, not compiled, not run on hardware
//
#include <hip/hip_runtime.h>
#include <hip/hip_bf16.h>
#include <math.h>

// Problem constants (match reference)
#define NN   60000
#define EE   300000
#define AAT  8000
#define GG   256
#define DD   128
#define HHD  8
#define CCH  16
#define LLAY 3
#define SBFD 112
#define RBFD 16

typedef __attribute__((ext_vector_type(16))) _Float16 v16h;
typedef __attribute__((ext_vector_type(8)))  _Float16 v8h;
typedef __attribute__((ext_vector_type(8)))  float    v8f;

__device__ __forceinline__ float silu_f(float x) { return x / (1.f + __expf(-x)); }

// ---------------------------------------------------------------------------
// WMMA linear: Y[M,128] = post( (gather?X[gidx[m]]:X[m]) @ W[K,128] + b )
// Templated on K (128 / 112 / 16) so the chunk loop fully unrolls.
// Block = 256 threads (8 waves), 64-row x 128-col macro tile:
//   - rowIdx + X tile (64 x KP f16) + W panel transposed (128 x KP f16) in LDS
//   - wave w owns 16-col tile n0 = w*16, holds 4 accumulators (4 row tiles)
//   - per 32-K chunk: B frag = 2x ds_load_b128, reused by 4 WMMAs
// ---------------------------------------------------------------------------
template <int K>
__global__ __launch_bounds__(256) void k_linear(
    const float* __restrict__ X, const float* __restrict__ W,
    const float* __restrict__ bias, const float* __restrict__ residual,
    const int* __restrict__ gidx, float* __restrict__ Y,
    int M, int act)
{
  constexpr int KP = (K + 31) & ~31;          // padded K (128,128,32)
  constexpr int LS = KP + 8;                  // f16 row stride (16B-aligned: LS*2 %16==0)
  __shared__ _Float16 Xh[64][LS];
  __shared__ _Float16 Wt[128][LS];            // transposed: Wt[n][k]
  __shared__ int rowIdx[64];

  const int tid = threadIdx.x;
  const int m0  = blockIdx.x * 64;

  if (tid < 64) {
    long row = m0 + tid;
    if (row >= M) row = M - 1;
    rowIdx[tid] = gidx ? gidx[row] : (int)row;
  }
  __syncthreads();

  // Stage X tile (64 x KP) as f16, coalesced float4 global reads.
  for (int t = tid * 4; t < 64 * KP; t += 1024) {
    int r = t / KP, c = t % KP;
    float4 xv = make_float4(0.f, 0.f, 0.f, 0.f);
    if (c < K) xv = *(const float4*)&X[(long)rowIdx[r] * K + c];
    Xh[r][c + 0] = (_Float16)xv.x; Xh[r][c + 1] = (_Float16)xv.y;
    Xh[r][c + 2] = (_Float16)xv.z; Xh[r][c + 3] = (_Float16)xv.w;
  }
  // Stage W panel (K x 128) transposed to Wt[n][k], f16.
  for (int t = tid * 4; t < KP * 128; t += 1024) {
    int k = t >> 7, n = t & 127;
    float4 wv = make_float4(0.f, 0.f, 0.f, 0.f);
    if (k < K) wv = *(const float4*)&W[(long)k * DD + n];
    Wt[n + 0][k] = (_Float16)wv.x; Wt[n + 1][k] = (_Float16)wv.y;
    Wt[n + 2][k] = (_Float16)wv.z; Wt[n + 3][k] = (_Float16)wv.w;
  }
  __syncthreads();

  const int lane  = tid & 31;
  const int wave  = tid >> 5;
  const int nloc  = wave * 16 + (lane & 15);        // output column
  const int kHalf = (lane < 16) ? 0 : 16;           // unused; B also uses kb
  const int kb    = (lane < 16) ? 0 : 8;            // A: lane-half K offset
  const int mrow  = lane & 15;

  v8f acc0 = {}, acc1 = {}, acc2 = {}, acc3 = {};
  #pragma unroll
  for (int kc = 0; kc < KP; kc += 32) {
    // B fragment: b[0..7] <-> K = kc+kb.., b[8..15] <-> K = kc+16+kb.., N = nloc  (2x b128 LDS reads)
    v8h b0 = *(const v8h*)&Wt[nloc][kc + kb];
    v8h b1 = *(const v8h*)&Wt[nloc][kc + 16 + kb];
    v16h bf = __builtin_shufflevector(b0, b1, 0,1,2,3,4,5,6,7,8,9,10,11,12,13,14,15);
    #pragma unroll
    for (int rt = 0; rt < 4; ++rt) {
      // A fragment: a[0..7] <-> K = kc+kb.., a[8..15] <-> K = kc+16+kb..
      v8h a0 = *(const v8h*)&Xh[rt * 16 + mrow][kc + kb];
      v8h a1 = *(const v8h*)&Xh[rt * 16 + mrow][kc + 16 + kb];
      v16h af = __builtin_shufflevector(a0, a1, 0,1,2,3,4,5,6,7,8,9,10,11,12,13,14,15);
      v8f c = (rt == 0) ? acc0 : (rt == 1) ? acc1 : (rt == 2) ? acc2 : acc3;
      c = __builtin_amdgcn_wmma_f32_16x16x32_f16(false, af, false, bf, (short)0, c,
                                                 false, false);
      if (rt == 0) acc0 = c; else if (rt == 1) acc1 = c;
      else if (rt == 2) acc2 = c; else acc3 = c;
    }
  }

  // Epilogue: C/D layout M = r + 8*(lane>=16), N = lane&15 (per 16x16 tile)
  const float bv = bias ? bias[nloc] : 0.f;
  #pragma unroll
  for (int rt = 0; rt < 4; ++rt) {
    if (m0 + rt * 16 >= M) break;                   // uniform per wave (M%16==0)
    v8f c = (rt == 0) ? acc0 : (rt == 1) ? acc1 : (rt == 2) ? acc2 : acc3;
    const long rbase = m0 + rt * 16 + ((lane < 16) ? 0 : 8);
    #pragma unroll
    for (int r = 0; r < 8; ++r) {
      long row = rbase + r;
      float val = c[r] + bv;
      if (act) val = silu_f(val);
      if (residual) val += residual[row * DD + nloc];
      Y[row * DD + nloc] = val;
    }
  }
}

static void launch_linear(const float* X, const float* W, const float* bias,
                          const float* residual, const int* gidx, float* Y,
                          int M, int K, int act, hipStream_t stream) {
  dim3 grid((M + 63) / 64, 1);
  if (K == 128)
    k_linear<128><<<grid, 256, 0, stream>>>(X, W, bias, residual, gidx, Y, M, act);
  else if (K == 112)
    k_linear<112><<<grid, 256, 0, stream>>>(X, W, bias, residual, gidx, Y, M, act);
  else
    k_linear<16><<<grid, 256, 0, stream>>>(X, W, bias, residual, gidx, Y, M, act);
}

// ---------------------------------------------------------------------------
// Elementwise / scatter helpers
// ---------------------------------------------------------------------------
__global__ void k_fill(float* p, long n, float v) {
  long i = (long)blockIdx.x * 256 + threadIdx.x;
  if (i < n) p[i] = v;
}
__global__ void k_copy(float* d, const float* s, long n) {
  long i = (long)blockIdx.x * 256 + threadIdx.x;
  if (i < n) d[i] = s[i];
}
__global__ void k_mul(float* d, const float* a, const float* b, long n) {
  long i = (long)blockIdx.x * 256 + threadIdx.x;
  if (i < n) d[i] = a[i] * b[i];
}
__global__ void k_scatter_add(float* dst, const float* src, const int* idx,
                              int rows, int cols) {
  long i = (long)blockIdx.x * 256 + threadIdx.x;
  if (i < (long)rows * cols) {
    int r = (int)(i / cols), c = (int)(i % cols);
    atomicAdd(&dst[(long)idx[r] * cols + c], src[i]);
  }
}

__device__ __forceinline__ void atomicMaxF(float* addr, float val) {
  int* ia = (int*)addr;
  int old = __float_as_int(*addr);
  while (__int_as_float(old) < val) {
    int assumed = old;
    old = atomicCAS(ia, assumed, __float_as_int(val));
    if (old == assumed) break;
  }
}

// ---------------------------------------------------------------------------
// Attention: per-edge logits, segment max, segment sum, message scatter
// ---------------------------------------------------------------------------
__global__ void k_logits(const float* __restrict__ q, const float* __restrict__ k,
                         const float* __restrict__ ek, const int* __restrict__ ei,
                         float* __restrict__ lg, float* __restrict__ nmax) {
  long i = (long)blockIdx.x * 256 + threadIdx.x;
  if (i >= (long)EE * HHD) return;
  int e = (int)(i / HHD), h = (int)(i % HHD);
  int s = ei[e], d = ei[EE + e];
  float acc = 0.f;
  #pragma unroll
  for (int c = 0; c < CCH; ++c) {
    int off = h * CCH + c;
    acc += q[(long)d * DD + off] * (k[(long)s * DD + off] + ek[(long)e * DD + off]);
  }
  acc *= 0.25f;                          // 1/sqrt(C), C=16
  lg[i] = acc;
  atomicMaxF(&nmax[(long)d * HHD + h], acc);
}

__global__ void k_expsum(float* __restrict__ lg, const float* __restrict__ nmax,
                         const int* __restrict__ ei, float* __restrict__ nsum) {
  long i = (long)blockIdx.x * 256 + threadIdx.x;
  if (i >= (long)EE * HHD) return;
  int e = (int)(i / HHD), h = (int)(i % HHD);
  int d = ei[EE + e];
  float ex = __expf(lg[i] - nmax[(long)d * HHD + h]);
  lg[i] = ex;
  atomicAdd(&nsum[(long)d * HHD + h], ex);
}

__global__ void k_message(const float* __restrict__ v, const float* __restrict__ ek,
                          const float* __restrict__ sw, const float* __restrict__ ex,
                          const float* __restrict__ nsum, const int* __restrict__ ei,
                          float* __restrict__ hout) {
  long i = (long)blockIdx.x * 256 + threadIdx.x;
  if (i >= (long)EE * DD) return;
  int e = (int)(i / DD), d = (int)(i % DD);
  int h = d >> 4;
  int s = ei[e], t = ei[EE + e];
  float alpha = ex[(long)e * HHD + h] / (nsum[(long)t * HHD + h] + 1e-16f);
  float msg = (v[(long)s * DD + d] + ek[i]) * alpha * sw[i];
  atomicAdd(&hout[(long)t * DD + d], msg);
}

// ---------------------------------------------------------------------------
// Graph norm (batch-aware LayerNorm, affine=False)
// ---------------------------------------------------------------------------
__global__ void k_gn_sum(const float* __restrict__ h, const int* __restrict__ batch,
                         float* __restrict__ gsum, float* __restrict__ gcnt) {
  int nidx = blockIdx.x * 256 + threadIdx.x;
  if (nidx >= NN) return;
  float s = 0.f;
  for (int d = 0; d < DD; ++d) s += h[(long)nidx * DD + d];
  int b = batch[nidx];
  atomicAdd(&gsum[b], s);
  atomicAdd(&gcnt[b], (float)DD);
}
__global__ void k_gn_mean(float* gsum, const float* gcnt) {
  int g = blockIdx.x * 256 + threadIdx.x;
  if (g < GG) gsum[g] = (gcnt[g] > 0.f) ? gsum[g] / gcnt[g] : 0.f;
}
__global__ void k_gn_var(const float* __restrict__ h, const int* __restrict__ batch,
                         const float* __restrict__ mean, float* __restrict__ gvar) {
  int nidx = blockIdx.x * 256 + threadIdx.x;
  if (nidx >= NN) return;
  float m = mean[batch[nidx]], s = 0.f;
  for (int d = 0; d < DD; ++d) {
    float t = h[(long)nidx * DD + d] - m;
    s += t * t;
  }
  atomicAdd(&gvar[batch[nidx]], s);
}
__global__ void k_gn_rstd(float* gvar, const float* gcnt) {
  int g = blockIdx.x * 256 + threadIdx.x;
  if (g < GG) gvar[g] = (gcnt[g] > 0.f) ? rsqrtf(gvar[g] / gcnt[g] + 1e-8f) : 0.f;
}
__global__ void k_gn_apply(float* __restrict__ h, const int* __restrict__ batch,
                           const float* __restrict__ mean, const float* __restrict__ rstd) {
  long i = (long)blockIdx.x * 256 + threadIdx.x;
  if (i >= (long)NN * DD) return;
  int b = batch[i / DD];
  h[i] = (h[i] - mean[b]) * rstd[b];
}

// ---------------------------------------------------------------------------
// Readout final projection (Dout=1) and graph reduction
// ---------------------------------------------------------------------------
__global__ void k_dot1(const float* __restrict__ a, const float* __restrict__ w3,
                       const float* __restrict__ b3, float* __restrict__ res) {
  int r = blockIdx.x * 256 + threadIdx.x;
  if (r >= AAT) return;
  float acc = b3[0];
  for (int d = 0; d < DD; ++d) acc += a[(long)r * DD + d] * w3[d];
  res[r] += acc;
}
__global__ void k_graph_out(const float* __restrict__ res, const int* __restrict__ ab,
                            float* __restrict__ gout) {
  int a = blockIdx.x * 256 + threadIdx.x;
  if (a >= AAT) return;
  atomicAdd(&gout[ab[a]], res[a] * (1.f / (float)LLAY));
}

// ---------------------------------------------------------------------------
extern "C" void kernel_launch(void* const* d_in, const int* in_sizes, int n_in,
                              void* d_out, int out_size, void* d_ws, size_t ws_size,
                              hipStream_t stream) {
  const float* x          = (const float*)d_in[0];
  const float* node_rbf   = (const float*)d_in[1];
  const float* edge_sbf   = (const float*)d_in[2];
  const int*   edge_index = (const int*)d_in[3];
  const int*   pair_idx   = (const int*)d_in[4];
  const int*   eidx0      = (const int*)d_in[5];
  const int*   atom_batch = (const int*)d_in[6];
  const int*   batch      = (const int*)d_in[7];
  const float* ew1 = (const float*)d_in[8];  const float* eb1 = (const float*)d_in[9];
  const float* ew2 = (const float*)d_in[10]; const float* eb2 = (const float*)d_in[11];
  const float* wq  = (const float*)d_in[12]; const float* wk  = (const float*)d_in[13];
  const float* wv  = (const float*)d_in[14]; const float* we  = (const float*)d_in[15];
  const float* wsbf = (const float*)d_in[16]; const float* bsbf = (const float*)d_in[17];
  const float* wrbf = (const float*)d_in[18];
  const float* dw  = (const float*)d_in[19]; const float* db  = (const float*)d_in[20];
  const float* bfw = (const float*)d_in[21]; const float* bfb = (const float*)d_in[22];
  const float* afw = (const float*)d_in[23]; const float* afb = (const float*)d_in[24];
  const float* rwrbf = (const float*)d_in[25];
  const float* rw1 = (const float*)d_in[26]; const float* rb1 = (const float*)d_in[27];
  const float* rw2 = (const float*)d_in[28]; const float* rb2 = (const float*)d_in[29];
  const float* rw3 = (const float*)d_in[30]; const float* rb3 = (const float*)d_in[31];
  float* gout = (float*)d_out;

  // Workspace carve (~540 MB total)
  char* p = (char*)d_ws;
  auto alloc = [&](size_t bytes) {
    void* r = (void*)p;
    p += (bytes + 255) & ~(size_t)255;
    return (float*)r;
  };
  const size_t NB = (size_t)NN * DD * 4, EB = (size_t)EE * DD * 4, AB = (size_t)AAT * DD * 4;
  float* f_out = alloc(NB); float* f_q  = alloc(NB); float* f_k  = alloc(NB);
  float* f_v   = alloc(NB); float* f_h  = alloc(NB); float* f_t1 = alloc(NB);
  float* f_t2  = alloc(NB);
  float* e_b1  = alloc(EB); float* e_b2 = alloc(EB);
  float* e_lg  = alloc((size_t)EE * HHD * 4);
  float* n_max = alloc((size_t)NN * HHD * 4);
  float* n_sum = alloc((size_t)NN * HHD * 4);
  float* a_rep = alloc(AB); float* a_t1 = alloc(AB); float* a_t2 = alloc(AB);
  float* a_res = alloc((size_t)AAT * 4);
  float* g_cnt = alloc((size_t)GG * 4);
  float* g_sum = alloc((size_t)GG * 4);
  float* g_var = alloc((size_t)GG * 4);
  (void)ws_size; (void)in_sizes; (void)n_in; (void)out_size;

  auto g1 = [](long n) { return dim3((unsigned)((n + 255) / 256)); };
#define LIN(X, W, B, RES, GI, Y, M, K, ACT) \
  launch_linear((X), (W), (B), (RES), (GI), (Y), (M), (K), (ACT), stream)

  const long ND = (long)NN * DD, ED = (long)EE * DD, AD = (long)AAT * DD;
  const long EH = (long)EE * HHD, NH = (long)NN * HHD;

  k_fill<<<g1(AAT), 256, 0, stream>>>(a_res, AAT, 0.f);
  k_copy<<<g1(ND), 256, 0, stream>>>(f_out, x, ND);

  // ---- readout(0, x) ----
  LIN(node_rbf, rwrbf, nullptr, nullptr, nullptr, f_t1, NN, RBFD, 0);
  k_mul<<<g1(ND), 256, 0, stream>>>(f_t2, f_out, f_t1, ND);
  k_fill<<<g1(AD), 256, 0, stream>>>(a_rep, AD, 0.f);
  k_scatter_add<<<g1(ND), 256, 0, stream>>>(a_rep, f_t2, eidx0, NN, DD);
  LIN(a_rep, rw1, rb1, nullptr, nullptr, a_t1, AAT, DD, 1);
  LIN(a_t1, rw2, rb2, nullptr, nullptr, a_t2, AAT, DD, 1);
  k_dot1<<<g1(AAT), 256, 0, stream>>>(a_t2, rw3, rb3, a_res);

  for (int i = 0; i < LLAY; ++i) {
    const long WD = (long)i * DD * DD;
    // edge feature path: gather atoms_rep -> 2-layer MLP -> ek
    k_fill<<<g1(AD), 256, 0, stream>>>(a_rep, AD, 0.f);
    k_scatter_add<<<g1(ND), 256, 0, stream>>>(a_rep, f_out, eidx0, NN, DD);
    LIN(a_rep, ew1 + WD, eb1 + (long)i * DD, nullptr, pair_idx, e_b2, EE, DD, 1);
    LIN(e_b2, ew2 + WD, eb2 + (long)i * DD, nullptr, nullptr, e_b1, EE, DD, 0);
    // q,k,v
    LIN(f_out, wq + WD, nullptr, nullptr, nullptr, f_q, NN, DD, 0);
    LIN(f_out, wk + WD, nullptr, nullptr, nullptr, f_k, NN, DD, 0);
    LIN(f_out, wv + WD, nullptr, nullptr, nullptr, f_v, NN, DD, 0);
    // ek, sw
    LIN(e_b1, we + WD, nullptr, nullptr, nullptr, e_b2, EE, DD, 0);
    LIN(edge_sbf, wsbf + (long)i * SBFD * DD, bsbf + (long)i * DD,
        nullptr, nullptr, e_b1, EE, SBFD, 0);
    // segment softmax over dst
    k_fill<<<g1(NH), 256, 0, stream>>>(n_max, NH, -1e30f);
    k_logits<<<g1(EH), 256, 0, stream>>>(f_q, f_k, e_b2, edge_index, e_lg, n_max);
    k_fill<<<g1(NH), 256, 0, stream>>>(n_sum, NH, 0.f);
    k_expsum<<<g1(EH), 256, 0, stream>>>(e_lg, n_max, edge_index, n_sum);
    // messages
    k_fill<<<g1(ND), 256, 0, stream>>>(f_h, ND, 0.f);
    k_message<<<g1(ED), 256, 0, stream>>>(f_v, e_b2, e_b1, e_lg, n_sum, edge_index, f_h);
    // rbf gate
    LIN(node_rbf, wrbf + (long)i * RBFD * DD, nullptr, nullptr, nullptr, f_t1, NN, RBFD, 0);
    k_mul<<<g1(ND), 256, 0, stream>>>(f_t2, f_h, f_t1, ND);
    // graph norm on f_t2
    k_fill<<<g1(GG), 256, 0, stream>>>(g_cnt, GG, 0.f);
    k_fill<<<g1(GG), 256, 0, stream>>>(g_sum, GG, 0.f);
    k_fill<<<g1(GG), 256, 0, stream>>>(g_var, GG, 0.f);
    k_gn_sum<<<g1(NN), 256, 0, stream>>>(f_t2, batch, g_sum, g_cnt);
    k_gn_mean<<<1, 256, 0, stream>>>(g_sum, g_cnt);
    k_gn_var<<<g1(NN), 256, 0, stream>>>(f_t2, batch, g_sum, g_var);
    k_gn_rstd<<<1, 256, 0, stream>>>(g_var, g_cnt);
    k_gn_apply<<<g1(ND), 256, 0, stream>>>(f_t2, batch, g_sum, g_var);
    // bf residual block
    LIN(f_t2, bfw + (long)(i * 2 + 0) * DD * DD, bfb + (long)(i * 2 + 0) * DD,
        nullptr, nullptr, f_t1, NN, DD, 1);
    LIN(f_t1, bfw + (long)(i * 2 + 1) * DD * DD, bfb + (long)(i * 2 + 1) * DD,
        f_t2, nullptr, f_h, NN, DD, 1);
    // dense + res0 (f_out)
    LIN(f_h, dw + WD, db + (long)i * DD, f_out, nullptr, f_t1, NN, DD, 1);
    // af residual block 1
    LIN(f_t1, afw + (long)(i * 4 + 0) * DD * DD, afb + (long)(i * 4 + 0) * DD,
        nullptr, nullptr, f_t2, NN, DD, 1);
    LIN(f_t2, afw + (long)(i * 4 + 1) * DD * DD, afb + (long)(i * 4 + 1) * DD,
        f_t1, nullptr, f_h, NN, DD, 1);
    // af residual block 2 -> new f_out
    LIN(f_h, afw + (long)(i * 4 + 2) * DD * DD, afb + (long)(i * 4 + 2) * DD,
        nullptr, nullptr, f_t2, NN, DD, 1);
    LIN(f_t2, afw + (long)(i * 4 + 3) * DD * DD, afb + (long)(i * 4 + 3) * DD,
        f_h, nullptr, f_out, NN, DD, 1);
    // readout(i+1, f_out)
    const long RL = (long)(i + 1);
    LIN(node_rbf, rwrbf + RL * RBFD * DD, nullptr, nullptr, nullptr, f_t1, NN, RBFD, 0);
    k_mul<<<g1(ND), 256, 0, stream>>>(f_t2, f_out, f_t1, ND);
    k_fill<<<g1(AD), 256, 0, stream>>>(a_rep, AD, 0.f);
    k_scatter_add<<<g1(ND), 256, 0, stream>>>(a_rep, f_t2, eidx0, NN, DD);
    LIN(a_rep, rw1 + RL * DD * DD, rb1 + RL * DD, nullptr, nullptr, a_t1, AAT, DD, 1);
    LIN(a_t1, rw2 + RL * DD * DD, rb2 + RL * DD, nullptr, nullptr, a_t2, AAT, DD, 1);
    k_dot1<<<g1(AAT), 256, 0, stream>>>(a_t2, rw3 + RL * DD, rb3 + RL, a_res);
  }

  // final graph reduction (includes /L)
  k_fill<<<g1(GG), 256, 0, stream>>>(gout, GG, 0.f);
  k_graph_out<<<g1(AAT), 256, 0, stream>>>(a_res, atom_batch, gout);
#undef LIN
}
